// Block_66322884984977
// MI455X (gfx1250) — matrix operand order repaired and checked by reading, not compiled, not run on hardware
//
#include <hip/hip_runtime.h>
#include <math.h>
#include <stdint.h>

// ---------------------------------------------------------------------------
// Types
// ---------------------------------------------------------------------------
typedef __bf16 bf16;
typedef __attribute__((ext_vector_type(16))) __bf16 v16bf;
typedef __attribute__((ext_vector_type(8)))  __bf16 bf16x8;
typedef __attribute__((ext_vector_type(8)))  float  v8f;
typedef __attribute__((ext_vector_type(4)))  float  f32x4;
typedef __attribute__((ext_vector_type(4)))  unsigned int u32x4;
typedef __attribute__((ext_vector_type(4)))  int    i32x4;
typedef __attribute__((ext_vector_type(8)))  int    i32x8;

#define DEVINL __device__ __forceinline__

#if __has_builtin(__builtin_amdgcn_tensor_load_to_lds) && \
    __has_builtin(__builtin_amdgcn_s_wait_tensorcnt)
#define HAVE_TDM 1
#endif

static constexpr int Bn = 4;
static constexpr int Tn = 1024;
static constexpr int Cn = 1024;
static constexpr int Hn = 16;     // heads
static constexpr int Dn = 64;     // head dim
static constexpr int Mn = Bn * Tn;   // 4096 rows
static constexpr int Fn = 4 * Cn;    // 4096 mlp hidden

// ---------------------------------------------------------------------------
// Helpers
// ---------------------------------------------------------------------------
DEVINL bf16 f2bf(float f) {
  unsigned u = __builtin_bit_cast(unsigned, f);
  unsigned r = (u + 0x7FFFu + ((u >> 16) & 1u)) >> 16;   // round-to-nearest-even
  return __builtin_bit_cast(bf16, (unsigned short)r);
}

DEVINL v8f wmma_bf16(v16bf a, v16bf b, v8f c) {
  // D = A(16x32) * B(32x16) + C, f32 accumulate
  return __builtin_amdgcn_wmma_f32_16x16x32_bf16(
      /*neg_a=*/false, a, /*neg_b=*/false, b,
      /*c_mod=*/(short)0, c, /*reuse_a=*/false, /*reuse_b=*/false);
}

// Wave-local LDS producer->consumer ordering (cross-lane via LDS):
// CDNA5 split counters: wait DScnt==0, fenced against compiler reordering.
DEVINL void wave_lds_sync() {
  __builtin_amdgcn_wave_barrier();
  asm volatile("s_wait_dscnt 0" ::: "memory");
  __builtin_amdgcn_wave_barrier();
}

// A fragment: 16(M) x 32(K), A row-major with leading dim lda (elements).
// ISA layout (16-bit A 16x32): lane = M%16; half-wave 0 holds K 0..7 / 16..23,
// half-wave 1 holds K 8..15 / 24..31  -> two contiguous 16B loads per lane.
DEVINL v16bf load_a(const bf16* A, int lda, int m0, int k0) {
  int lane = threadIdx.x & 31;
  int m = lane & 15, hl = (lane >> 4) & 1;
  const bf16* p = A + (size_t)(m0 + m) * lda + k0 + hl * 8;
  bf16x8 lo = *(const bf16x8*)p;
  bf16x8 hi = *(const bf16x8*)(p + 16);
  v16bf r;
#pragma unroll
  for (int i = 0; i < 8; ++i) { r[i] = lo[i]; r[i + 8] = hi[i]; }
  return r;
}

// B fragment: 32(K) x 16(N), given B^T stored row-major [N x K] (ldb = K).
// ISA layout (16-bit B 32x16): lane = N%16; half-wave 0 holds K 0..7 / 16..23,
// half-wave 1 holds K 8..15 / 24..31 -> two contiguous 16B loads per lane.
DEVINL v16bf load_bt(const bf16* BT, int ldb, int n0, int k0) {
  int lane = threadIdx.x & 31;
  int n = lane & 15, hl = (lane >> 4) & 1;
  const bf16* p = BT + (size_t)(n0 + n) * ldb + k0 + hl * 8;
  bf16x8 lo = *(const bf16x8*)p;
  bf16x8 hi = *(const bf16x8*)(p + 16);
  v16bf r;
#pragma unroll
  for (int i = 0; i < 8; ++i) { r[i] = lo[i]; r[i + 8] = hi[i]; }
  return r;
}

#ifdef HAVE_TDM
// ---------------------------------------------------------------------------
// Tensor Data Mover: DMA a 2D bf16 tile (tile_rows x tile_k) from a row-major
// [rows x ldb] matrix into LDS (compacted row-major tile). D# per ISA 8.3/8.4:
//   group0: count=1 | lds_addr | global_addr(57b) | type=2
//   group1: data_size=1(2B), tensor_dim0=ldb, dim0_stride=ldb,
//           tile_dim0=tile_k, tile_dim1=tile_rows
// ---------------------------------------------------------------------------
DEVINL void tdm_load_tile(unsigned lds_addr, const bf16* gaddr, int ldb,
                          int tile_k, int tile_rows) {
  unsigned long long ga = (unsigned long long)(uintptr_t)gaddr;
  u32x4 g0;
  g0[0] = 1u;                                             // count=1, user desc
  g0[1] = lds_addr;                                       // LDS byte address
  g0[2] = (unsigned)ga;                                   // global addr [31:0]
  g0[3] = (unsigned)((ga >> 32) & 0x01FFFFFFu) | (2u << 30);  // [56:32] | type=2
  unsigned long long dim0 = (unsigned long long)(unsigned)ldb;
  unsigned long long dim1 = 0x7FFFFFFFull;                // "plenty" of rows
  unsigned long long st0  = (unsigned long long)(unsigned)ldb;
  i32x8 g1;
  g1[0] = (int)(1u << 16);                                // data_size = 2 bytes
  g1[1] = (int)(((unsigned)(dim0 & 0xFFFFu)) << 16);      // dim0[15:0] @ bits63:48
  g1[2] = (int)(((unsigned)((dim0 >> 16) & 0xFFFFu)) |    // dim0[31:16]
                (((unsigned)(dim1 & 0xFFFFu)) << 16));    // dim1[15:0]
  g1[3] = (int)(((unsigned)((dim1 >> 16) & 0xFFFFu)) |    // dim1[31:16]
                (((unsigned)tile_k) << 16));              // tile_dim0
  g1[4] = (int)((unsigned)tile_rows);                     // tile_dim1 (dim2=0)
  g1[5] = (int)(unsigned)(st0 & 0xFFFFFFFFull);           // stride0[31:0]
  g1[6] = (int)(unsigned)((st0 >> 32) & 0xFFFFu);         // stride0[47:32]
  g1[7] = 0;                                              // stride1 = 0
  i32x4 z4 = {};
#if defined(__clang_major__) && (__clang_major__ >= 23)
  i32x8 z8 = {};
  __builtin_amdgcn_tensor_load_to_lds(g0, g1, z4, z4, z8, 0);
#else
  __builtin_amdgcn_tensor_load_to_lds(g0, g1, z4, z4, 0);
#endif
}
#endif  // HAVE_TDM

// ---------------------------------------------------------------------------
// Weight convert + transpose:  W[K x N] fp32  ->  W^T[N x K] bf16
// ---------------------------------------------------------------------------
__global__ void k_cvt_transpose(const float* __restrict__ W, bf16* __restrict__ WT,
                                int K, int N) {
  size_t i = (size_t)blockIdx.x * blockDim.x + threadIdx.x;
  size_t total = (size_t)K * N;
  if (i >= total) return;
  int k = (int)(i / N), n = (int)(i % N);
  WT[(size_t)n * K + k] = f2bf(W[i]);
}

__global__ void k_cvt(const float* __restrict__ X, bf16* __restrict__ Y, size_t n) {
  size_t i = (size_t)blockIdx.x * blockDim.x + threadIdx.x;
  if (i < n) Y[i] = f2bf(X[i]);
}

// V (B,T,C=H*d) bf16 -> V^T (B,H,d,T) bf16 so P@V B-fragments are contiguous
__global__ void k_vtrans(const bf16* __restrict__ Vin, bf16* __restrict__ VT) {
  size_t i = (size_t)blockIdx.x * blockDim.x + threadIdx.x;
  if (i >= (size_t)Mn * Cn) return;
  int c = (int)(i % Cn);
  size_t bt = i / Cn;
  int t = (int)(bt % Tn);
  int b = (int)(bt / Tn);
  int h = c >> 6, f = c & 63;
  VT[(((size_t)b * Hn + h) * Dn + f) * Tn + t] = Vin[i];
}

// ---------------------------------------------------------------------------
// LayerNorm: fp32 row -> bf16 normalized row.  One 256-thread block per row.
// ---------------------------------------------------------------------------
__global__ __launch_bounds__(256) void k_layernorm(
    const float* __restrict__ X, const float* __restrict__ g,
    const float* __restrict__ be, bf16* __restrict__ out) {
  int row = blockIdx.x;
  int tid = threadIdx.x;
  const float* xr = X + (size_t)row * Cn;
  f32x4 v4 = ((const f32x4*)xr)[tid];           // 256*4 = 1024 = Cn
  float s  = v4[0] + v4[1] + v4[2] + v4[3];
  float s2 = v4[0]*v4[0] + v4[1]*v4[1] + v4[2]*v4[2] + v4[3]*v4[3];
#pragma unroll
  for (int off = 1; off < 32; off <<= 1) {
    s  += __shfl_xor(s,  off, 32);
    s2 += __shfl_xor(s2, off, 32);
  }
  __shared__ float ws[8], ws2[8];
  if ((tid & 31) == 0) { ws[tid >> 5] = s; ws2[tid >> 5] = s2; }
  __syncthreads();
  float ts = 0.f, ts2 = 0.f;
#pragma unroll
  for (int i = 0; i < 8; ++i) { ts += ws[i]; ts2 += ws2[i]; }
  float mu   = ts * (1.0f / Cn);
  float var  = ts2 * (1.0f / Cn) - mu * mu;
  float rstd = rsqrtf(var + 1e-5f);
  bf16* orow = out + (size_t)row * Cn;
#pragma unroll
  for (int j = 0; j < 4; ++j) {
    int c = tid * 4 + j;
    orow[c] = f2bf((v4[j] - mu) * rstd * g[c] + be[c]);
  }
}

// ---------------------------------------------------------------------------
// GEMM: out = act(A[MxK](bf16) @ B^T[NxK](bf16)^T + bias) (+ residual)
// 256 threads = 8 waves; wave w computes a 32x64 strip (8 wmma accumulators).
// The 64x64 B^T panel is staged in LDS by the Tensor Data Mover (double-
// buffered on TENSORcnt); A streams from global with prefetch.
// Grid: (N/64, M/256)
// ---------------------------------------------------------------------------
template <bool GELU>
__global__ __launch_bounds__(256) void k_gemm(
    const bf16* __restrict__ A, const bf16* __restrict__ BT,
    const float* __restrict__ bias, const float* __restrict__ residual,
    float* __restrict__ outF, bf16* __restrict__ outB,
    int M, int N, int Kd) {
  __shared__ __align__(16) bf16 btile[2][64 * 64];   // 2 x 8 KB panels
  int wave = threadIdx.x >> 5;
  int lane = threadIdx.x & 31;
  int col = lane & 15, hl = (lane >> 4) & 1;
  int m0 = blockIdx.y * 256 + wave * 32;
  int n0 = blockIdx.x * 64;

  v8f acc0[4] = {}, acc1[4] = {};

#ifdef HAVE_TDM
  if (wave == 0) {
    tdm_load_tile((unsigned)(uintptr_t)&btile[0][0],
                  BT + (size_t)n0 * Kd, Kd, 64, 64);
  }
  int buf = 0;
  for (int k0 = 0; k0 < Kd; k0 += 64) {
    if (wave == 0) {
      if (k0 + 64 < Kd) {
        tdm_load_tile((unsigned)(uintptr_t)&btile[buf ^ 1][0],
                      BT + (size_t)n0 * Kd + k0 + 64, Kd, 64, 64);
        __builtin_amdgcn_s_wait_tensorcnt((short)1);   // oldest (current) done
      } else {
        __builtin_amdgcn_s_wait_tensorcnt((short)0);
      }
    }
    __syncthreads();
    const bf16* bt = &btile[buf][0];
    // prefetch next A chunk (one row per lane covers the 32-row strip)
    if (k0 + 64 < Kd)
      __builtin_prefetch(A + (size_t)(m0 + lane) * Kd + k0 + 64, 0, 3);
#pragma unroll
    for (int ks = 0; ks < 2; ++ks) {
      int kk = k0 + ks * 32;
      v16bf a0 = load_a(A, Kd, m0, kk);
      v16bf a1 = load_a(A, Kd, m0 + 16, kk);
#pragma unroll
      for (int t = 0; t < 4; ++t) {
        v16bf b = load_bt(bt, 64, t * 16, ks * 32);
        acc0[t] = wmma_bf16(a0, b, acc0[t]);
        acc1[t] = wmma_bf16(a1, b, acc1[t]);
      }
    }
    __syncthreads();
    buf ^= 1;
  }
#else
  for (int k0 = 0; k0 < Kd; k0 += 64) {
    for (int i = threadIdx.x; i < 512; i += 256) {     // 64x64 panel, 8 elems/iter
      int rr = i >> 3, cc = (i & 7) * 8;
      *(bf16x8*)&btile[0][rr * 64 + cc] =
          *(const bf16x8*)&BT[(size_t)(n0 + rr) * Kd + k0 + cc];
    }
    __syncthreads();
    const bf16* bt = &btile[0][0];
#pragma unroll
    for (int ks = 0; ks < 2; ++ks) {
      int kk = k0 + ks * 32;
      v16bf a0 = load_a(A, Kd, m0, kk);
      v16bf a1 = load_a(A, Kd, m0 + 16, kk);
#pragma unroll
      for (int t = 0; t < 4; ++t) {
        v16bf b = load_bt(bt, 64, t * 16, ks * 32);
        acc0[t] = wmma_bf16(a0, b, acc0[t]);
        acc1[t] = wmma_bf16(a1, b, acc1[t]);
      }
    }
    __syncthreads();
  }
#endif

  // Epilogue. C/D layout: VGPR r, lane -> row = r + 8*hl, col = lane&15
#pragma unroll
  for (int t = 0; t < 4; ++t) {
    int n = n0 + t * 16 + col;
    float bv = bias ? bias[n] : 0.f;
#pragma unroll
    for (int r = 0; r < 8; ++r) {
#pragma unroll
      for (int s = 0; s < 2; ++s) {
        int row = m0 + s * 16 + r + hl * 8;
        float v = (s == 0 ? acc0[t][r] : acc1[t][r]) + bv;
        if (GELU) v = 0.5f * v * (1.f + erff(v * 0.70710678118654752f));
        if (residual) v += residual[(size_t)row * N + n];
        if (outF) outF[(size_t)row * N + n] = v;
        if (outB) outB[(size_t)row * N + n] = f2bf(v);
      }
    }
  }
}

// ---------------------------------------------------------------------------
// Flash attention. Q,K: (B,T,C) bf16; VT: (B,H,d,T) bf16; Y: (B,T,C) bf16.
// One wave per 16-query tile; 8 waves per block. Grid: (T/128, H, B).
// ---------------------------------------------------------------------------
__global__ __launch_bounds__(256) void k_attn(
    const bf16* __restrict__ Q, const bf16* __restrict__ Kmat,
    const bf16* __restrict__ VT, bf16* __restrict__ Y,
    int Tq, int Tk, int causal) {
  __shared__ __align__(16) bf16 pbuf[8][16 * 32];   // per-wave P tile (1 KB each)
  int wave = threadIdx.x >> 5;
  int lane = threadIdx.x & 31;
  int col = lane & 15, hl = (lane >> 4) & 1;
  int q0 = (blockIdx.x * 8 + wave) * 16;
  int h = blockIdx.y, b = blockIdx.z;

  const bf16* Qb = Q    + ((size_t)b * Tq) * Cn + h * Dn;
  const bf16* Kb = Kmat + ((size_t)b * Tk) * Cn + h * Dn;
  const bf16* Vb = VT   + (((size_t)b * Hn + h) * Dn) * (size_t)Tk;  // [feat][key]
  bf16*       Yb = Y    + ((size_t)b * Tq) * Cn + h * Dn;

  v16bf aQ0 = load_a(Qb, Cn, q0, 0);
  v16bf aQ1 = load_a(Qb, Cn, q0, 32);

  v8f o[4] = {};
  float mrow[8], lrow[8];
#pragma unroll
  for (int r = 0; r < 8; ++r) { mrow[r] = -1e30f; lrow[r] = 0.f; }

  const float scale = 0.125f;   // 1/sqrt(64)
  int send = causal ? (q0 + 16) : Tk;

  for (int s0 = 0; s0 < send; s0 += 32) {
    // ---- S = Q @ K^T for 16 queries x 32 keys (two 16x16 tiles) ----
    v8f sc0 = {}, sc1 = {};
    {
      v16bf bk = load_bt(Kb, Cn, s0, 0);       sc0 = wmma_bf16(aQ0, bk, sc0);
      bk       = load_bt(Kb, Cn, s0, 32);      sc0 = wmma_bf16(aQ1, bk, sc0);
      bk       = load_bt(Kb, Cn, s0 + 16, 0);  sc1 = wmma_bf16(aQ0, bk, sc1);
      bk       = load_bt(Kb, Cn, s0 + 16, 32); sc1 = wmma_bf16(aQ1, bk, sc1);
    }
    // ---- online softmax (row stats per r; rows split across half-waves) ----
    bf16* P = pbuf[wave];
#pragma unroll
    for (int r = 0; r < 8; ++r) {
      int qrow = q0 + r + hl * 8;
      int k0i = s0 + col, k1i = s0 + 16 + col;
      float x0 = sc0[r] * scale, x1 = sc1[r] * scale;
      if (k0i >= Tk || (causal && k0i > qrow)) x0 = -1e30f;
      if (k1i >= Tk || (causal && k1i > qrow)) x1 = -1e30f;
      float tmax = fmaxf(x0, x1);
#pragma unroll
      for (int off = 1; off < 16; off <<= 1) tmax = fmaxf(tmax, __shfl_xor(tmax, off, 16));
      float mnew = fmaxf(mrow[r], tmax);
      float p0 = __expf(x0 - mnew), p1 = __expf(x1 - mnew);
      float rs = p0 + p1;
#pragma unroll
      for (int off = 1; off < 16; off <<= 1) rs += __shfl_xor(rs, off, 16);
      float corr = __expf(mrow[r] - mnew);
      lrow[r] = lrow[r] * corr + rs;
      mrow[r] = mnew;
      o[0][r] *= corr; o[1][r] *= corr; o[2][r] *= corr; o[3][r] *= corr;
      int prow = r + hl * 8;
      P[prow * 32 + col]      = f2bf(p0);
      P[prow * 32 + 16 + col] = f2bf(p1);
    }
    wave_lds_sync();                       // LDS stores -> cross-lane loads
    v16bf aP = load_a(P, 32, 0, 0);        // P as A-fragment (16x32)
    wave_lds_sync();                       // keep next-iter stores after loads
    // ---- O += P @ V  (V^T rows are contiguous keys) ----
#pragma unroll
    for (int t = 0; t < 4; ++t) {
      v16bf bv = load_bt(Vb, Tk, t * 16, s0);
      o[t] = wmma_bf16(aP, bv, o[t]);
    }
  }
  // ---- normalize + store ----
#pragma unroll
  for (int r = 0; r < 8; ++r) {
    int row = q0 + r + hl * 8;
    float inv = 1.0f / lrow[r];
#pragma unroll
    for (int t = 0; t < 4; ++t) {
      Yb[(size_t)row * Cn + t * 16 + col] = f2bf(o[t][r] * inv);
    }
  }
}

// ---------------------------------------------------------------------------
// Host-side launcher
// ---------------------------------------------------------------------------
extern "C" void kernel_launch(void* const* d_in, const int* in_sizes, int n_in,
                              void* d_out, int out_size, void* d_ws, size_t ws_size,
                              hipStream_t stream) {
  const float* x       = (const float*)d_in[0];
  const float* context = (const float*)d_in[1];
  const float* ln1_g = (const float*)d_in[2];
  const float* ln1_b = (const float*)d_in[3];
  const float* ln2_g = (const float*)d_in[4];
  const float* ln2_b = (const float*)d_in[5];
  const float* sa_w[4] = { (const float*)d_in[6],  (const float*)d_in[8],
                           (const float*)d_in[10], (const float*)d_in[12] };
  const float* sa_bias[4] = { (const float*)d_in[7],  (const float*)d_in[9],
                              (const float*)d_in[11], (const float*)d_in[13] };
  const float* ca_w[4] = { (const float*)d_in[14], (const float*)d_in[16],
                           (const float*)d_in[18], (const float*)d_in[20] };
  const float* ca_bias[4] = { (const float*)d_in[15], (const float*)d_in[17],
                              (const float*)d_in[19], (const float*)d_in[21] };
  const float* mlp_w1 = (const float*)d_in[22];
  const float* mlp_b1 = (const float*)d_in[23];
  const float* mlp_w2 = (const float*)d_in[24];
  const float* mlp_b2 = (const float*)d_in[25];
  float* out = (float*)d_out;

  // ---- workspace bump allocator ----
  char* ws = (char*)d_ws;
  size_t off = 0;
  auto alloc = [&](size_t bytes) -> void* {
    void* p = ws + off;
    off = (off + bytes + 255) & ~(size_t)255;
    return p;
  };
  const size_t szCC  = (size_t)Cn * Cn * sizeof(bf16);
  const size_t szMC2 = (size_t)Mn * Cn * sizeof(bf16);
  const size_t szMC4 = (size_t)Mn * Cn * sizeof(float);

  bf16* wT[8];
  for (int i = 0; i < 8; ++i) wT[i] = (bf16*)alloc(szCC);   // sa q,k,v,o ; ca q,k,v,o
  bf16* w1t  = (bf16*)alloc((size_t)Cn * Fn * sizeof(bf16));
  bf16* w2t  = (bf16*)alloc((size_t)Fn * Cn * sizeof(bf16));
  bf16* lnb  = (bf16*)alloc(szMC2);
  bf16* ctxb = (bf16*)alloc(szMC2);
  bf16* qb   = (bf16*)alloc(szMC2);
  bf16* kb   = (bf16*)alloc(szMC2);
  bf16* vb   = (bf16*)alloc(szMC2);
  bf16* vtb  = (bf16*)alloc(szMC2);
  bf16* yb   = (bf16*)alloc(szMC2);
  float* x1  = (float*)alloc(szMC4);
  float* x2  = (float*)alloc(szMC4);
  bf16* h1   = (bf16*)alloc((size_t)Mn * Fn * sizeof(bf16));

  // ---- one-time converts ----
  {
    int nb = (Cn * Cn + 255) / 256;
    for (int i = 0; i < 4; ++i)
      k_cvt_transpose<<<nb, 256, 0, stream>>>(sa_w[i], wT[i], Cn, Cn);
    for (int i = 0; i < 4; ++i)
      k_cvt_transpose<<<nb, 256, 0, stream>>>(ca_w[i], wT[4 + i], Cn, Cn);
    k_cvt_transpose<<<(Cn * Fn + 255) / 256, 256, 0, stream>>>(mlp_w1, w1t, Cn, Fn);
    k_cvt_transpose<<<(Fn * Cn + 255) / 256, 256, 0, stream>>>(mlp_w2, w2t, Fn, Cn);
    k_cvt<<<((size_t)Mn * Cn + 255) / 256, 256, 0, stream>>>(context, ctxb, (size_t)Mn * Cn);
  }

  dim3 gC(Cn / 64, Mn / 256);   // GEMM grid for N=1024 (blocks: 256 rows x 64 cols)
  dim3 gF(Fn / 64, Mn / 256);   // GEMM grid for N=4096
  dim3 gA(Tn / 128, Hn, Bn);    // attention grid
  int nbMC = (Mn * Cn + 255) / 256;

  // ======== self-attention (causal) ========
  k_layernorm<<<Mn, 256, 0, stream>>>(x, ln1_g, ln1_b, lnb);
  k_gemm<false><<<gC, 256, 0, stream>>>(lnb, wT[0], sa_bias[0], nullptr, nullptr, qb, Mn, Cn, Cn);
  k_gemm<false><<<gC, 256, 0, stream>>>(lnb, wT[1], sa_bias[1], nullptr, nullptr, kb, Mn, Cn, Cn);
  k_gemm<false><<<gC, 256, 0, stream>>>(lnb, wT[2], sa_bias[2], nullptr, nullptr, vb, Mn, Cn, Cn);
  k_vtrans<<<nbMC, 256, 0, stream>>>(vb, vtb);
  k_attn<<<gA, 256, 0, stream>>>(qb, kb, vtb, yb, Tn, Tn, 1);
  k_gemm<false><<<gC, 256, 0, stream>>>(yb, wT[3], sa_bias[3], x, x1, nullptr, Mn, Cn, Cn);

  // ======== cross-attention (non-causal) ========
  k_layernorm<<<Mn, 256, 0, stream>>>(x1, ln2_g, ln2_b, lnb);
  k_gemm<false><<<gC, 256, 0, stream>>>(lnb,  wT[4], ca_bias[0], nullptr, nullptr, qb, Mn, Cn, Cn);
  k_gemm<false><<<gC, 256, 0, stream>>>(ctxb, wT[5], ca_bias[1], nullptr, nullptr, kb, Mn, Cn, Cn);
  k_gemm<false><<<gC, 256, 0, stream>>>(ctxb, wT[6], ca_bias[2], nullptr, nullptr, vb, Mn, Cn, Cn);
  k_vtrans<<<nbMC, 256, 0, stream>>>(vb, vtb);
  k_attn<<<gA, 256, 0, stream>>>(qb, kb, vtb, yb, Tn, Tn, 0);
  k_gemm<false><<<gC, 256, 0, stream>>>(yb, wT[7], ca_bias[3], x1, x2, nullptr, Mn, Cn, Cn);

  // ======== MLP (reuses ln2 params, exact GELU) ========
  k_layernorm<<<Mn, 256, 0, stream>>>(x2, ln2_g, ln2_b, lnb);
  k_gemm<true><<<gF, 256, 0, stream>>>(lnb, w1t, mlp_b1, nullptr, nullptr, h1, Mn, Fn, Cn);
  k_gemm<false><<<gC, 256, 0, stream>>>(h1, w2t, mlp_b2, x2, out, nullptr, Mn, Cn, Fn);

  (void)in_sizes; (void)n_in; (void)out_size; (void)ws_size;
}
